// ImageEncoderViT_14980845928833
// MI455X (gfx1250) — matrix operand order repaired and checked by reading, not compiled, not run on hardware
//
#include <hip/hip_runtime.h>
#include <math.h>

typedef unsigned short u16;
typedef unsigned int   u32;
typedef long long      ll;

typedef __attribute__((ext_vector_type(16))) __bf16 v16bf;
typedef __attribute__((ext_vector_type(8)))  float  v8f;

struct U2 { u32 x, y; };
struct U4 { u32 x, y, z, w; };
union FragBF { v16bf v; U4 q[2]; };

#define BM 128
#define BN 128
#define BK 32
#define LDT 40   // padded LDS row stride in halfs (80B: breaks bank conflicts, keeps 16B align)

__device__ __forceinline__ u16 f2bf(float f) {
  u32 u = __float_as_uint(f);
  u32 r = (u + 0x7FFFu + ((u >> 16) & 1u)) >> 16;  // round-to-nearest-even
  return (u16)r;
}
__device__ __forceinline__ float bf2f(u16 h) {
  return __uint_as_float(((u32)h) << 16);
}
__device__ __forceinline__ float wred_sum(float v) {
#pragma unroll
  for (int m = 16; m > 0; m >>= 1) v += __shfl_xor(v, m, 32);
  return v;
}
__device__ __forceinline__ float wred_max(float v) {
#pragma unroll
  for (int m = 16; m > 0; m >>= 1) v = fmaxf(v, __shfl_xor(v, m, 32));
  return v;
}
__device__ __forceinline__ float gelu_exact(float x) {
  return 0.5f * x * (1.0f + erff(x * 0.7071067811865475f));
}

// CDNA5 async global->LDS copy (ASYNCcnt-tracked, no VGPR round trip).
// VDST operand = wave-relative LDS byte address = low 32 bits of flat LDS pointer.
__device__ __forceinline__ void async_ld_b128(u16* lds, const u16* g) {
  u32 l = (u32)(size_t)lds;
  asm volatile("global_load_async_to_lds_b128 %0, %1, off"
               :: "v"(l), "v"(g) : "memory");
}
__device__ __forceinline__ void wait_async0() {
  asm volatile("s_wait_asynccnt 0x0" ::: "memory");
}

// ---------------------------------------------------------------------------
// Generic batched GEMM: C[z] = act( alpha * A[z] (MxK, bf16 rowmajor)
//                                  * B[z]^T (B stored N x K bf16 rowmajor)
//                                  + bias + addsrc[z] )
// out fp32 (outF) or bf16 (outB). Requires K % 4 == 0.
// 256 threads (8 waves), block tile 128x128, wave tile 32x64 (2x4 WMMA frags).
// Double-buffered LDS; bulk tiles staged with global_load_async_to_lds_b128.
// ---------------------------------------------------------------------------
__global__ __launch_bounds__(256)
void k_gemm(const u16* __restrict__ A, ll sA,
            const u16* __restrict__ B, ll sB,
            float alpha,
            const float* __restrict__ bias,
            const float* __restrict__ addsrc, ll sAdd,
            float* __restrict__ outF, u16* __restrict__ outB, ll sC,
            int M, int N, int K, int act)
{
  int z = blockIdx.z;
  A += (ll)z * sA;
  B += (ll)z * sB;
  if (addsrc) addsrc += (ll)z * sAdd;
  ll coff = (ll)z * sC;

  __shared__ __align__(16) u16 As[2][BM * LDT];
  __shared__ __align__(16) u16 Bs[2][BN * LDT];

  int tid  = threadIdx.x;
  int lane = tid & 31;
  int wave = tid >> 5;
  int wm   = wave & 3;        // 4 waves along M (32 rows each)
  int wn   = wave >> 2;       // 2 waves along N (64 cols each)
  int l16  = lane & 15;
  int lhi  = lane >> 4;
  int bm0  = blockIdx.x * BM;
  int bn0  = blockIdx.y * BN;

  // staging map: thread -> (row, 16-half segment); 2 b128 per matrix per tile
  int ldrow = tid >> 1;
  int ldseg = (tid & 1) * 16;

  bool k8     = (K & 7) == 0;                 // 16B row alignment for b128 async
  bool asyncA = ((bm0 + BM) <= M) && k8;
  bool asyncB = ((bn0 + BN) <= N) && k8;
  bool arok   = (bm0 + ldrow) < M;
  bool brok   = (bn0 + ldrow) < N;
  const u16* Abase = A + (ll)(bm0 + ldrow) * K + ldseg;
  const u16* Bbase = B + (ll)(bn0 + ldrow) * K + ldseg;

  int kTiles = (K + BK - 1) / BK;

  auto stage = [&](int kt, int buf) {
    int k0 = kt * BK;
    bool kfull = (k0 + BK) <= K;
    u16* da = &As[buf][ldrow * LDT + ldseg];
    u16* db = &Bs[buf][ldrow * LDT + ldseg];
    const u16* sa = Abase + k0;
    const u16* sb = Bbase + k0;
    if (asyncA && kfull) {
      async_ld_b128(da, sa);
      async_ld_b128(da + 8, sa + 8);
    } else {
      U2 v[4];
#pragma unroll
      for (int g = 0; g < 4; ++g) {
        v[g].x = 0u; v[g].y = 0u;
        if (arok && (k0 + ldseg + g * 4) < K) v[g] = *(const U2*)(sa + g * 4);
      }
#pragma unroll
      for (int g = 0; g < 4; ++g) *(U2*)(da + g * 4) = v[g];
    }
    if (asyncB && kfull) {
      async_ld_b128(db, sb);
      async_ld_b128(db + 8, sb + 8);
    } else {
      U2 v[4];
#pragma unroll
      for (int g = 0; g < 4; ++g) {
        v[g].x = 0u; v[g].y = 0u;
        if (brok && (k0 + ldseg + g * 4) < K) v[g] = *(const U2*)(sb + g * 4);
      }
#pragma unroll
      for (int g = 0; g < 4; ++g) *(U2*)(db + g * 4) = v[g];
    }
  };

  v8f acc[2][4] = {};

  stage(0, 0);
  for (int kt = 0; kt < kTiles; ++kt) {
    int cur = kt & 1;
    wait_async0();        // our async copies for tile kt are done
    __syncthreads();      // everyone's copies done; prior reads of other buf done
    if (kt + 1 < kTiles) stage(kt + 1, cur ^ 1);   // overlap with compute below

    FragBF af[2];
#pragma unroll
    for (int f = 0; f < 2; ++f) {
      // A 16x32 bf16 frag: lane m=l16; lhi=0 -> K0..7 & K16..23; lhi=1 -> K8..15 & K24..31
      const u16* p = &As[cur][(wm * 32 + f * 16 + l16) * LDT + lhi * 8];
      af[f].q[0] = *(const U4*)(p);
      af[f].q[1] = *(const U4*)(p + 16);
    }
    FragBF bfm[4];
#pragma unroll
    for (int g = 0; g < 4; ++g) {
      // B 32x16 bf16 frag: lane col n=l16; lhi=0 -> K0..7 & K16..23; lhi=1 -> K8..15 & K24..31
      const u16* p = &Bs[cur][(wn * 64 + g * 16 + l16) * LDT + lhi * 8];
      bfm[g].q[0] = *(const U4*)(p);
      bfm[g].q[1] = *(const U4*)(p + 16);
    }
#pragma unroll
    for (int f = 0; f < 2; ++f)
#pragma unroll
      for (int g = 0; g < 4; ++g)
        acc[f][g] = __builtin_amdgcn_wmma_f32_16x16x32_bf16(
            false, af[f].v, false, bfm[g].v, (short)0, acc[f][g], false, false);
  }

  // --- epilogue: C/D layout row = 8*lhi + r, col = l16 ---
#pragma unroll
  for (int f = 0; f < 2; ++f) {
#pragma unroll
    for (int g = 0; g < 4; ++g) {
      int col = bn0 + wn * 64 + g * 16 + l16;
      if (col >= N) continue;
      float bia = bias ? bias[col] : 0.0f;
#pragma unroll
      for (int r = 0; r < 8; ++r) {
        int row = bm0 + wm * 32 + f * 16 + lhi * 8 + r;
        if (row >= M) continue;
        float v = alpha * acc[f][g][r] + bia;
        if (addsrc) v += addsrc[(ll)row * N + col];
        if (act == 1) v = gelu_exact(v);
        ll idx = coff + (ll)row * N + col;
        if (outF) outF[idx] = v;
        else      outB[idx] = f2bf(v);
      }
    }
  }
}

// ---------------------------------------------------------------------------
// Elementwise / helper kernels
// ---------------------------------------------------------------------------
__global__ void k_wtrans(const float* __restrict__ W, u16* __restrict__ WT,
                         int K, int N, ll total) {
  ll i = (ll)blockIdx.x * blockDim.x + threadIdx.x;
  if (i >= total) return;
  int k = (int)(i / N), n = (int)(i % N);
  WT[(ll)n * K + k] = f2bf(W[i]);
}

__global__ void k_f2b(const float* __restrict__ in, u16* __restrict__ out, ll total) {
  ll i = (ll)blockIdx.x * blockDim.x + threadIdx.x;
  if (i < total) out[i] = f2bf(in[i]);
}

__global__ void k_add(float* __restrict__ a, const float* __restrict__ b, ll total) {
  ll i = (ll)blockIdx.x * blockDim.x + threadIdx.x;
  if (i < total) a[i] += b[i];
}

__global__ void k_layernorm(const float* __restrict__ X,
                            const float* __restrict__ w, const float* __restrict__ b,
                            float eps, int M, int C,
                            u16* __restrict__ outB, float* __restrict__ outF,
                            int transposeOut) {
  int row  = blockIdx.x * (blockDim.x >> 5) + (threadIdx.x >> 5);
  int lane = threadIdx.x & 31;
  if (row >= M) return;
  const float* x = X + (ll)row * C;
  float s = 0.f, s2 = 0.f;
  for (int c = lane; c < C; c += 32) { float v = x[c]; s += v; s2 += v * v; }
  s = wred_sum(s); s2 = wred_sum(s2);
  float mean = s / (float)C;
  float var  = s2 / (float)C - mean * mean;
  float inv  = rsqrtf(var + eps);
  for (int c = lane; c < C; c += 32) {
    float v = (x[c] - mean) * inv * w[c] + b[c];
    if (outF) {
      if (transposeOut) outF[(ll)c * M + row] = v;
      else              outF[(ll)row * C + c] = v;
    }
    if (outB) outB[(ll)row * C + c] = f2bf(v);
  }
}

// rel-pos: out[bh][q][kk] = dot64(q_vec, rel[coord - kk + size - 1])
__global__ void k_relpos(const u16* __restrict__ Q, const float* __restrict__ rel,
                         float* __restrict__ out, ll total,
                         int Nq, int W, int size, int modeW) {
  ll i = (ll)blockIdx.x * blockDim.x + threadIdx.x;
  if (i >= total) return;
  int kk = (int)(i % size);
  ll r = i / size;                 // bh*Nq + q
  int q = (int)(r % Nq);
  int coord = modeW ? (q % W) : (q / W);
  const u16*   qp = Q   + r * 64;
  const float* rp = rel + (ll)(coord - kk + size - 1) * 64;
  float acc = 0.f;
#pragma unroll 8
  for (int c = 0; c < 64; ++c) acc += bf2f(qp[c]) * rp[c];
  out[i] = acc;
}

// softmax over keys with fused rel-h/rel-w add; one wave per (bh, q) row
__global__ void k_softmax(const float* __restrict__ S,
                          const float* __restrict__ relh, const float* __restrict__ relw,
                          u16* __restrict__ P, int N, int H, int W, ll nrows) {
  ll row = (ll)blockIdx.x * (blockDim.x >> 5) + (threadIdx.x >> 5);
  int lane = threadIdx.x & 31;
  if (row >= nrows) return;
  const float* s  = S    + row * N;
  const float* rh = relh + row * H;
  const float* rw = relw + row * W;
  float mx = -3.0e38f;
  for (int k = lane; k < N; k += 32)
    mx = fmaxf(mx, s[k] + rh[k / W] + rw[k % W]);
  mx = wred_max(mx);
  float sum = 0.f;
  for (int k = lane; k < N; k += 32)
    sum += __expf(s[k] + rh[k / W] + rw[k % W] - mx);
  sum = wred_sum(sum);
  float inv = 1.0f / sum;
  u16* p = P + row * N;
  for (int k = lane; k < N; k += 32)
    p[k] = f2bf(__expf(s[k] + rh[k / W] + rw[k % W] - mx) * inv);
}

// qkv (rows x 2304 fp32) -> q,k bf16 [BH][N][64], vT bf16 [BH][64][N]
__global__ void k_split_qkv(const float* __restrict__ qkv,
                            u16* __restrict__ q, u16* __restrict__ k, u16* __restrict__ vt,
                            int nb, int N, ll total) {
  ll i = (ll)blockIdx.x * blockDim.x + threadIdx.x;
  if (i >= total) return;
  int d = (int)(i & 63);
  ll t = i >> 6;
  int h = (int)(t % 12); t /= 12;
  int n = (int)(t % N);
  int b = (int)(t / N);
  ll rbase = ((ll)(b * N + n)) * 2304 + h * 64 + d;
  ll bh = (ll)(b * 12 + h);
  q[(bh * N + n) * 64 + d]        = f2bf(qkv[rbase]);
  k[(bh * N + n) * 64 + d]        = f2bf(qkv[rbase + 768]);
  vt[bh * 64 * N + (ll)d * N + n] = f2bf(qkv[rbase + 1536]);
}

// attn_out [BH][N][64] fp32 -> merged bf16 [nb*N][768]
__global__ void k_merge_heads(const float* __restrict__ ao, u16* __restrict__ m,
                              int nb, int N, ll total) {
  ll i = (ll)blockIdx.x * blockDim.x + threadIdx.x;
  if (i >= total) return;
  int c = (int)(i % 768);
  ll r = i / 768;
  int n = (int)(r % N);
  int b = (int)(r / N);
  int h = c >> 6, d = c & 63;
  m[i] = f2bf(ao[(((ll)(b * 12 + h)) * N + n) * 64 + d]);
}

// 32x32x768 bf16 -> 9 windows of 14x14x768 bf16 (zero padded)
__global__ void k_winpart(const u16* __restrict__ t, u16* __restrict__ w, ll total) {
  ll i = (ll)blockIdx.x * blockDim.x + threadIdx.x;
  if (i >= total) return;
  int c = (int)(i % 768);
  ll r = i / 768;
  int win = (int)(r / 196), p = (int)(r % 196);
  int wy = p / 14, wx = p % 14;
  int y = (win / 3) * 14 + wy, x = (win % 3) * 14 + wx;
  w[i] = (y < 32 && x < 32) ? t[((ll)y * 32 + x) * 768 + c] : (u16)0;
}

// windows (fp32, 9x196x768) -> h += unpartitioned
__global__ void k_winunpart_add(const float* __restrict__ g, float* __restrict__ h, ll total) {
  ll i = (ll)blockIdx.x * blockDim.x + threadIdx.x;
  if (i >= total) return;
  int c = (int)(i % 768);
  ll r = i / 768;
  int y = (int)(r / 32), x = (int)(r % 32);
  int win = (y / 14) * 3 + (x / 14);
  int p = (y % 14) * 14 + (x % 14);
  h[i] += g[(((ll)win) * 196 + p) * 768 + c];
}

// NCHW 3x512x512 -> im2col bf16 (1024 x 768), K index = (ph*16+pw)*3+c (HWIO)
__global__ void k_im2col_patch(const float* __restrict__ x, u16* __restrict__ A, ll total) {
  ll i = (ll)blockIdx.x * blockDim.x + threadIdx.x;
  if (i >= total) return;
  int kidx = (int)(i % 768);
  ll tok = i / 768;
  int gy = (int)(tok / 32), gx = (int)(tok % 32);
  int c = kidx % 3, pp = kidx / 3;
  int ph = pp / 16, pw = pp % 16;
  A[i] = f2bf(x[((ll)c * 512 + (gy * 16 + ph)) * 512 + (gx * 16 + pw)]);
}

// 32x32x256 bf16 -> im2col bf16 (1024 x 2304) for 3x3 SAME, K = (ky*3+kx)*256+c
__global__ void k_im2col3(const u16* __restrict__ in, u16* __restrict__ A, ll total) {
  ll i = (ll)blockIdx.x * blockDim.x + threadIdx.x;
  if (i >= total) return;
  int kidx = (int)(i % 2304);
  ll tok = i / 2304;
  int y = (int)(tok / 32), x = (int)(tok % 32);
  int c = kidx % 256, kp = kidx / 256;
  int yy = y + kp / 3 - 1, xx = x + kp % 3 - 1;
  A[i] = (yy >= 0 && yy < 32 && xx >= 0 && xx < 32)
             ? in[((ll)yy * 32 + xx) * 256 + c] : (u16)0;
}

// ---------------------------------------------------------------------------
// Host orchestration
// ---------------------------------------------------------------------------
#define EW(total) ((unsigned)(((total) + 255) / 256))

static void gemm(hipStream_t st, const u16* A, ll sA, const u16* B, ll sB,
                 float alpha, const float* bias, const float* addsrc, ll sAdd,
                 float* outF, u16* outB, ll sC, int M, int N, int K, int act, int batch) {
  dim3 g((unsigned)((M + BM - 1) / BM), (unsigned)((N + BN - 1) / BN), (unsigned)batch);
  k_gemm<<<g, 256, 0, st>>>(A, sA, B, sB, alpha, bias, addsrc, sAdd,
                            outF, outB, sC, M, N, K, act);
}

extern "C" void kernel_launch(void* const* d_in, const int* in_sizes, int n_in,
                              void* d_out, int out_size, void* d_ws, size_t ws_size,
                              hipStream_t stream) {
  const float* x          = (const float*)d_in[0];
  const float* patch_w    = (const float*)d_in[1];
  const float* patch_b    = (const float*)d_in[2];
  const float* pos_embed  = (const float*)d_in[3];
  const float* ln1_w      = (const float*)d_in[4];
  const float* ln1_b      = (const float*)d_in[5];
  const float* qkv_w      = (const float*)d_in[6];
  const float* qkv_b      = (const float*)d_in[7];
  const float* proj_w     = (const float*)d_in[8];
  const float* proj_b     = (const float*)d_in[9];
  const float* ln2_w      = (const float*)d_in[10];
  const float* ln2_b      = (const float*)d_in[11];
  const float* fc1_w      = (const float*)d_in[12];
  const float* fc1_b      = (const float*)d_in[13];
  const float* fc2_w      = (const float*)d_in[14];
  const float* fc2_b      = (const float*)d_in[15];
  const float* rel_h_win  = (const float*)d_in[16];
  const float* rel_w_win  = (const float*)d_in[17];
  const float* rel_h_glob = (const float*)d_in[18];
  const float* rel_w_glob = (const float*)d_in[19];
  const float* neck_w1    = (const float*)d_in[20];
  const float* neck_ln1_w = (const float*)d_in[21];
  const float* neck_ln1_b = (const float*)d_in[22];
  const float* neck_w2    = (const float*)d_in[23];
  const float* neck_ln2_w = (const float*)d_in[24];
  const float* neck_ln2_b = (const float*)d_in[25];

  const float SCALE = 0.125f;  // 64^-0.5

  // ---- workspace layout ----
  char* p = (char*)d_ws;
  auto alloc = [&](size_t bytes) -> void* {
    void* r = (void*)p;
    p += (bytes + 255) & ~(size_t)255;
    return r;
  };
  const ll RMAX = 9LL * 196;              // 1764 window rows
  const ll BHN_MAX = 108LL * 196 * 64;    // 1,354,752

  u16*   WT     = (u16*)  alloc(3072LL * 768 * 2);
  float* h      = (float*)alloc(1024LL * 768 * 4);
  u16*   tb     = (u16*)  alloc(1024LL * 768 * 2);
  u16*   winb   = (u16*)  alloc(RMAX * 768 * 2);
  float* qkvf   = (float*)alloc(RMAX * 2304 * 4);
  u16*   qbuf   = (u16*)  alloc(BHN_MAX * 2);
  u16*   kbuf   = (u16*)  alloc(BHN_MAX * 2);
  u16*   vtbuf  = (u16*)  alloc(BHN_MAX * 2);
  float* scores = (float*)alloc(12LL * 1024 * 1024 * 4);
  u16*   probs  = (u16*)  alloc(12LL * 1024 * 1024 * 2);
  float* relh   = (float*)alloc(12LL * 1024 * 32 * 4);
  float* relw   = (float*)alloc(12LL * 1024 * 32 * 4);
  float* attnout= (float*)alloc(BHN_MAX * 4);
  u16*   merged = (u16*)  alloc(RMAX * 768 * 2);
  float* gtmp   = (float*)alloc(RMAX * 768 * 4);
  u16*   hid    = (u16*)  alloc(1024LL * 3072 * 2);
  u16*   icol   = (u16*)  alloc(1024LL * 2304 * 2);
  float* nk1    = (float*)alloc(1024LL * 256 * 4);
  u16*   nk1b   = (u16*)  alloc(1024LL * 256 * 2);
  float* nk2    = (float*)alloc(1024LL * 256 * 4);

  // ---- patch embed: im2col + GEMM (+ bias, + pos_embed) ----
  {
    ll tot = 1024LL * 768;
    k_im2col_patch<<<EW(tot), 256, 0, stream>>>(x, icol, tot);
    k_wtrans<<<EW(768LL * 768), 256, 0, stream>>>(patch_w, WT, 768, 768, 768LL * 768);
    gemm(stream, icol, 0, WT, 0, 1.0f, patch_b, pos_embed, 0,
         h, nullptr, 0, 1024, 768, 768, 0, 1);
  }

  // ---- transformer blocks ----
  int wi = 0, gi = 0;
  for (int i = 0; i < 12; ++i) {
    bool isGlobal = (i == 2 || i == 5 || i == 8 || i == 11);
    // LN1 -> bf16
    k_layernorm<<<(1024 + 7) / 8, 256, 0, stream>>>(
        h, ln1_w + i * 768, ln1_b + i * 768, 1e-5f, 1024, 768, tb, nullptr, 0);

    const u16* abuf; int rows, nb, N, H, W;
    const float *rhT, *rwT;
    if (isGlobal) {
      abuf = tb; rows = 1024; nb = 1; N = 1024; H = W = 32;
      rhT = rel_h_glob + (ll)gi * 63 * 64;
      rwT = rel_w_glob + (ll)gi * 63 * 64;
      ++gi;
    } else {
      ll tot = (ll)RMAX * 768;
      k_winpart<<<EW(tot), 256, 0, stream>>>(tb, winb, tot);
      abuf = winb; rows = 9 * 196; nb = 9; N = 196; H = W = 14;
      rhT = rel_h_win + (ll)wi * 27 * 64;
      rwT = rel_w_win + (ll)wi * 27 * 64;
      ++wi;
    }
    int BH = nb * 12;

    // qkv GEMM
    k_wtrans<<<EW(768LL * 2304), 256, 0, stream>>>(
        qkv_w + (ll)i * 768 * 2304, WT, 768, 2304, 768LL * 2304);
    gemm(stream, abuf, 0, WT, 0, 1.0f, qkv_b + (ll)i * 2304, nullptr, 0,
         qkvf, nullptr, 0, rows, 2304, 768, 0, 1);

    // split q/k/vT
    {
      ll tot = (ll)rows * 768;  // rows * 12 heads * 64
      k_split_qkv<<<EW(tot), 256, 0, stream>>>(qkvf, qbuf, kbuf, vtbuf, nb, N, tot);
    }

    // scores = SCALE * q k^T   (batched over BH)
    gemm(stream, qbuf, (ll)N * 64, kbuf, (ll)N * 64, SCALE, nullptr, nullptr, 0,
         scores, nullptr, (ll)N * N, N, N, 64, 0, BH);

    // rel-pos contributions
    {
      ll th = (ll)BH * N * H;
      ll tw = (ll)BH * N * W;
      k_relpos<<<EW(th), 256, 0, stream>>>(qbuf, rhT, relh, th, N, W, H, 0);
      k_relpos<<<EW(tw), 256, 0, stream>>>(qbuf, rwT, relw, tw, N, W, W, 1);
    }

    // softmax (fused rel add) -> probs bf16
    {
      ll nrows = (ll)BH * N;
      k_softmax<<<(unsigned)((nrows + 7) / 8), 256, 0, stream>>>(
          scores, relh, relw, probs, N, H, W, nrows);
    }

    // attn_out = probs @ v  (B = vT, N x 64 output, K = N keys)
    gemm(stream, probs, (ll)N * N, vtbuf, (ll)64 * N, 1.0f, nullptr, nullptr, 0,
         attnout, nullptr, (ll)N * 64, N, 64, N, 0, BH);

    // merge heads -> bf16
    {
      ll tot = (ll)rows * 768;
      k_merge_heads<<<EW(tot), 256, 0, stream>>>(attnout, merged, nb, N, tot);
    }

    // proj GEMM
    k_wtrans<<<EW(768LL * 768), 256, 0, stream>>>(
        proj_w + (ll)i * 768 * 768, WT, 768, 768, 768LL * 768);
    gemm(stream, merged, 0, WT, 0, 1.0f, proj_b + (ll)i * 768, nullptr, 0,
         gtmp, nullptr, 0, rows, 768, 768, 0, 1);

    // residual add (with window unpartition for windowed blocks)
    if (isGlobal) {
      k_add<<<EW(1024LL * 768), 256, 0, stream>>>(h, gtmp, 1024LL * 768);
    } else {
      ll tot = 1024LL * 768;
      k_winunpart_add<<<EW(tot), 256, 0, stream>>>(gtmp, h, tot);
    }

    // MLP: LN2 -> fc1 (+gelu, bf16 out) -> fc2 (+residual)
    k_layernorm<<<(1024 + 7) / 8, 256, 0, stream>>>(
        h, ln2_w + i * 768, ln2_b + i * 768, 1e-5f, 1024, 768, tb, nullptr, 0);
    k_wtrans<<<EW(768LL * 3072), 256, 0, stream>>>(
        fc1_w + (ll)i * 768 * 3072, WT, 768, 3072, 768LL * 3072);
    gemm(stream, tb, 0, WT, 0, 1.0f, fc1_b + (ll)i * 3072, nullptr, 0,
         nullptr, hid, 0, 1024, 3072, 768, 1, 1);
    k_wtrans<<<EW(3072LL * 768), 256, 0, stream>>>(
        fc2_w + (ll)i * 3072 * 768, WT, 3072, 768, 3072LL * 768);
    gemm(stream, hid, 0, WT, 0, 1.0f, fc2_b + (ll)i * 768, h, 0,
         h, nullptr, 0, 1024, 768, 3072, 0, 1);
  }

  // ---- neck ----
  k_f2b<<<EW(1024LL * 768), 256, 0, stream>>>(h, tb, 1024LL * 768);
  k_wtrans<<<EW(768LL * 256), 256, 0, stream>>>(neck_w1, WT, 768, 256, 768LL * 256);
  gemm(stream, tb, 0, WT, 0, 1.0f, nullptr, nullptr, 0,
       nk1, nullptr, 0, 1024, 256, 768, 0, 1);
  k_layernorm<<<(1024 + 7) / 8, 256, 0, stream>>>(
      nk1, neck_ln1_w, neck_ln1_b, 1e-6f, 1024, 256, nk1b, nullptr, 0);
  {
    ll tot = 1024LL * 2304;
    k_im2col3<<<EW(tot), 256, 0, stream>>>(nk1b, icol, tot);
  }
  k_wtrans<<<EW(2304LL * 256), 256, 0, stream>>>(neck_w2, WT, 2304, 256, 2304LL * 256);
  gemm(stream, icol, 0, WT, 0, 1.0f, nullptr, nullptr, 0,
       nk2, nullptr, 0, 1024, 256, 2304, 0, 1);
  // final LN writes NCHW-transposed straight to d_out
  k_layernorm<<<(1024 + 7) / 8, 256, 0, stream>>>(
      nk2, neck_ln2_w, neck_ln2_b, 1e-6f, 1024, 256, nullptr, (float*)d_out, 1);
}
